// RBF_Conv2d_54305566490847
// MI455X (gfx1250) — matrix operand order repaired and checked by reading, not compiled, not run on hardware
//
#include <hip/hip_runtime.h>
#include <hip/hip_bf16.h>

typedef __attribute__((ext_vector_type(16))) _Float16 v16h;
typedef __attribute__((ext_vector_type(8)))  _Float16 v8h;
typedef __attribute__((ext_vector_type(8)))  float    v8f;

#define B_   16
#define C_   64
#define HW   56
#define HWP  58          // padded 56 + 2
#define PL   3364        // 58*58 padded plane
#define O_   128
#define CKK  576         // 64*3*3
#define L_   3136        // 56*56
#define KT   18          // 576 / 32

// workspace layout (bytes)
#define WS_WF  0          // 147456 B : f16 swizzled weights
#define WS_X2  147456     // 200704 B : f32 x2[b][l]
#define WS_W2  348160     // 512 B    : f32 w2[o]
#define WS_XH  348672     // 6889472 B: f16 zero-padded x [b][c][58][58]

// ---------------------------------------------------------------------------
// Kernel 1: swizzle weight (O x 576, f32) into WMMA A-fragment layout (f16).
// ---------------------------------------------------------------------------
__global__ __launch_bounds__(256) void prep_w_swizzle(const float* __restrict__ w,
                                                      _Float16* __restrict__ wf) {
    int idx  = blockIdx.x * 256 + threadIdx.x;     // 0 .. 73727
    int j    = idx & 15;
    int lane = (idx >> 4) & 31;
    int grp  = idx >> 9;                           // og*18 + kt
    int og   = grp / KT;
    int kt   = grp - og * KT;
    int hi   = lane >> 4;
    int kmap = hi * 16 + j;
    int o    = og * 16 + (lane & 15);
    int k    = kt * 32 + kmap;
    wf[idx]  = (_Float16)w[o * CKK + k];
}

// ---------------------------------------------------------------------------
// Kernel 2: w2[o] = sum_k w[o][k]^2
// ---------------------------------------------------------------------------
__global__ void prep_w2(const float* __restrict__ w, float* __restrict__ w2) {
    int o = threadIdx.x;
    if (o < O_) {
        const float* row = w + o * CKK;
        float s = 0.f;
        for (int k = 0; k < CKK; ++k) { float v = row[k]; s += v * v; }
        w2[o] = s;
    }
}

// ---------------------------------------------------------------------------
// Kernel 3: x2[b][l] = sum over C and 3x3 (zero-padded) window of x^2 (f32).
// ---------------------------------------------------------------------------
__global__ __launch_bounds__(256) void prep_x2(const float* __restrict__ x,
                                               float* __restrict__ x2) {
    int gid = blockIdx.x * 256 + threadIdx.x;      // 0 .. 50175
    int b   = gid / L_;
    int l   = gid - b * L_;
    int oh  = l / HW;
    int ow  = l - oh * HW;
    float s = 0.f;
    for (int c = 0; c < C_; ++c) {
        const float* plane = x + ((b * C_ + c) * HW) * HW;
        #pragma unroll
        for (int kh = 0; kh < 3; ++kh) {
            int ih = oh + kh - 1;
            if ((unsigned)ih < (unsigned)HW) {
                const float* rowp = plane + ih * HW;
                #pragma unroll
                for (int kw = 0; kw < 3; ++kw) {
                    int iw = ow + kw - 1;
                    if ((unsigned)iw < (unsigned)HW) { float v = rowp[iw]; s += v * v; }
                }
            }
        }
    }
    x2[gid] = s;
}

// ---------------------------------------------------------------------------
// Kernel 4: build zero-padded f16 copy of x: xh[b][c][58][58].
// ---------------------------------------------------------------------------
__global__ __launch_bounds__(256) void prep_pad(const float* __restrict__ x,
                                                _Float16* __restrict__ xh) {
    int gid = blockIdx.x * 256 + threadIdx.x;      // 0 .. 3444735
    int bc  = gid / PL;                            // b*64 + c
    int p   = gid - bc * PL;
    int ih  = p / HWP;
    int iw  = p - ih * HWP;
    _Float16 v = (_Float16)0.f;
    if ((unsigned)(ih - 1) < (unsigned)HW && (unsigned)(iw - 1) < (unsigned)HW)
        v = (_Float16)x[(bc * HW + (ih - 1)) * HW + (iw - 1)];
    xh[gid] = v;
}

// ---------------------------------------------------------------------------
// Fill helper (round-4 addressing form: SADDR base + 32-bit index + SGPR
// constant add per element). kt and KR literal -> soff is a compile-time
// constant; compiler keeps xh in SADDR and does one v_add per element.
// ---------------------------------------------------------------------------
__device__ __forceinline__ void fill_tile(const _Float16* __restrict__ xh,
                                          int pre, v16h& regs, int kt, int KR) {
    #pragma unroll
    for (int e = 0; e < 16; ++e) {
        int kk   = kt * 32 + KR * 16 + e;      // compile-time after inline
        int c    = kk / 9;
        int r9   = kk - c * 9;
        int kh   = r9 / 3;
        int kw   = r9 - kh * 3;
        int soff = c * PL + kh * HWP + kw;     // constant
        regs[e]  = xh[pre + soff];             // global_load_u16 (scale_offset)
    }
}

// ---------------------------------------------------------------------------
// Main GEMM: block = (batch b, 64-column L tile), 128 threads = 4 waves.
// Wave wv: 32 O-rows (two A fragments) x 64 cols = 8 WMMA/step, each LDS B
// fragment feeding two v_wmma_f32_16x16x32_f16. K loop (18 x 32) fully
// unrolled; branchless padded-f16 im2col; LDS double buffered, one
// barrier/step; next-step globals issued before the barrier.
// ---------------------------------------------------------------------------
__global__ __launch_bounds__(128) void rbf_main(const _Float16* __restrict__ xh,
                                                const _Float16* __restrict__ wf,
                                                const float* __restrict__ x2,
                                                const float* __restrict__ w2,
                                                float* __restrict__ out) {
    const int ltile = blockIdx.x;          // 0..48
    const int b     = blockIdx.y;          // 0..15
    const int l0    = ltile * 64;
    const int t     = threadIdx.x;
    const int lane  = t & 31;
    const int wv    = t >> 5;              // 0..3

    __shared__ __align__(32) _Float16 Pl[2][4 * 32 * 16];   // 2 x 4KB

    // Fill mapping: thread owns column lcol, k = krow0*16 + e (e = 0..15):
    // one full B-fragment LDS row -> one contiguous 32B LDS store.
    const int lcol  = t & 63;
    const int krow0 = __builtin_amdgcn_readfirstlane(t >> 6);   // 0 or 1
    const int l     = l0 + lcol;
    const int oh    = l / HW;
    const int ow    = l - oh * HW;
    const int fcol  = lcol >> 4;
    const int ncol  = lcol & 15;

    const int pre    = b * C_ * PL + oh * HWP + ow;   // per-lane base index
    const int ldsrow = fcol * 32 + krow0 * 16 + ncol;

    // per-wave A-fragment base pointers (kt offset becomes an immediate)
    const int og0 = wv * 2, og1 = wv * 2 + 1;
    const _Float16* wfA0 = wf + (og0 * KT * 32 + lane) * 16;
    const _Float16* wfA1 = wf + (og1 * KT * 32 + lane) * 16;

    v8f acc[2][4];
    #pragma unroll
    for (int i = 0; i < 2; ++i)
        #pragma unroll
        for (int f = 0; f < 4; ++f) acc[i][f] = (v8f){};

    v16h regs;
    if (krow0 == 0) fill_tile(xh, pre, regs, 0, 0);
    else            fill_tile(xh, pre, regs, 0, 1);
    v16h a0 = *(const v16h*)(wfA0);
    v16h a1 = *(const v16h*)(wfA1);

    #pragma unroll
    for (int kt = 0; kt < KT; ++kt) {
        const int p = kt & 1;
        *(v16h*)(&Pl[p][ldsrow * 16]) = regs;      // one 32B LDS store

        v16h a0n = a0, a1n = a1;
        if (kt + 1 < KT) {                         // prefetch before barrier
            if (krow0 == 0) fill_tile(xh, pre, regs, kt + 1, 0);
            else            fill_tile(xh, pre, regs, kt + 1, 1);
            a0n = *(const v16h*)(wfA0 + (kt + 1) * 512);
            a1n = *(const v16h*)(wfA1 + (kt + 1) * 512);
        }

        __syncthreads();

        #pragma unroll
        for (int f = 0; f < 4; ++f) {
            v16h bm = *(const v16h*)(&Pl[p][(f * 32 + lane) * 16]);
            acc[0][f] = __builtin_amdgcn_wmma_f32_16x16x32_f16(false, a0, false, bm, (short)0, acc[0][f], false, false);
            acc[1][f] = __builtin_amdgcn_wmma_f32_16x16x32_f16(false, a1, false, bm, (short)0, acc[1][f], false, false);
        }

        a0 = a0n; a1 = a1n;                        // renamed away by unroll
    }

    // Epilogue: d2 = max(x2[l] + w2[o] - 2*xw, 0)
    // C/D layout: VGPR r -> M = r + (lane>>4)*8, column = lane&15
    // Output is streamed once -> nontemporal stores (don't pollute L2).
    const int nlo = lane & 15;
    const int hi  = lane >> 4;
    #pragma unroll
    for (int f = 0; f < 4; ++f) {
        int col  = l0 + f * 16 + nlo;
        float xv = x2[b * L_ + col];
        #pragma unroll
        for (int i = 0; i < 2; ++i) {
            #pragma unroll
            for (int r = 0; r < 8; ++r) {
                int o   = (wv * 2 + i) * 16 + hi * 8 + r;
                float d = xv + w2[o] - 2.0f * acc[i][f][r];
                __builtin_nontemporal_store(fmaxf(d, 0.0f),
                                            &out[(b * O_ + o) * L_ + col]);
            }
        }
    }
}

// ---------------------------------------------------------------------------
// Host launcher
// ---------------------------------------------------------------------------
extern "C" void kernel_launch(void* const* d_in, const int* in_sizes, int n_in,
                              void* d_out, int out_size, void* d_ws, size_t ws_size,
                              hipStream_t stream) {
    const float* x = (const float*)d_in[0];     // (16, 64, 56, 56) f32
    const float* w = (const float*)d_in[1];     // (128, 576) f32
    float* out = (float*)d_out;                 // (16, 128, 56, 56) f32

    char* ws = (char*)d_ws;
    _Float16* wf = (_Float16*)(ws + WS_WF);
    float*    x2 = (float*)(ws + WS_X2);
    float*    w2 = (float*)(ws + WS_W2);
    _Float16* xh = (_Float16*)(ws + WS_XH);

    prep_w_swizzle<<<dim3(288), dim3(256), 0, stream>>>(w, wf);        // 73728
    prep_w2<<<dim3(1), dim3(128), 0, stream>>>(w, w2);
    prep_x2<<<dim3(196), dim3(256), 0, stream>>>(x, x2);               // 50176
    prep_pad<<<dim3(13456), dim3(256), 0, stream>>>(x, xh);            // 3444736

    dim3 grid(49, 16);                                                 // L x B
    rbf_main<<<grid, dim3(128), 0, stream>>>(xh, wf, x2, w2, out);
}
